// RWKV6_CrossAttBlock_62775241998770
// MI455X (gfx1250) — compile-verified, matrix operand order checked by reading, not run on hardware
//
#include <hip/hip_runtime.h>
#include <stdint.h>
#include <stddef.h>

// ---------------- problem constants ----------------
#define Bn 4
#define Tn 512
#define Dn 1024
#define Hn 16
#define DKn 64
#define DVn 64
#define Rn 32
#define GLRn 64
#define INTERn 3584
#define BTn (Bn*Tn)
#define EPSn 1e-5f

typedef __bf16 bf16_t;
typedef __attribute__((ext_vector_type(16))) __bf16 v16bf;
typedef __attribute__((ext_vector_type(8)))  __bf16 v8bf;
typedef __attribute__((ext_vector_type(8)))  float  v8f;

// ---------------- WMMA GEMM ----------------
// C[M,N] = epi( A[M,K] @ B + bias )
// A: bf16 row-major, lda.  B: BTR==0 -> bf16 [K,N] row-major (ldb);
//                          BTR==1 -> bf16 stored [N,K] (ldb) -> multiply by transpose.
//                                    (BTR==1 call sites always have N==1024: no guards)
// Epilogues: 0 = f32, 1 = +bias[n] f32, 2 = tanh -> bf16, 3 = sqrelu -> bf16
#define TILE_M 128
#define TILE_N 64
#define KSTEP  32
#define LDSROW 40   // 32 + 8 pad bf16; 80B row stride keeps 16B align, spreads banks

__device__ __forceinline__ void async_b128(uint32_t lds_off, uint64_t gaddr) {
  // GLOBAL_LOAD_ASYNC_TO_LDS_B128 (GV mode): memory -> LDS, tracked by ASYNCcnt
  asm volatile("global_load_async_to_lds_b128 %0, %1, off"
               :: "v"(lds_off), "v"(gaddr) : "memory");
}

__device__ __forceinline__ v16bf load_frag(const bf16_t* rowbase, int half) {
  v8bf lo = *(const v8bf*)(rowbase + half * 8);
  v8bf hi = *(const v8bf*)(rowbase + 16 + half * 8);
  return __builtin_shufflevector(lo, hi, 0,1,2,3,4,5,6,7,8,9,10,11,12,13,14,15);
}

template<int BTR>
__device__ __forceinline__ void stage_tile(const bf16_t* __restrict__ A, int lda,
                                           const bf16_t* __restrict__ Bm, int ldb,
                                           int m0, int n0, int N, int k0,
                                           bf16_t* Abuf, bf16_t* Bbuf, int tid)
{
  // A tile 128x32: 512 b128-chunks / 128 threads = 4 async per thread
  const uint32_t a_lds = (uint32_t)(uintptr_t)Abuf;
#pragma unroll
  for (int p = 0; p < 4; ++p) {
    int it  = tid + p * 128;
    int row = it >> 2, ch = it & 3;
    async_b128(a_lds + (uint32_t)(row * LDSROW + ch * 8) * 2,
               (uint64_t)(A + (size_t)(m0 + row) * lda + k0 + ch * 8));
  }
  if (BTR) {
    // B stored [N,K]: same contiguous-chunk shape, 64x32 -> 2 async per thread
    const uint32_t b_lds = (uint32_t)(uintptr_t)Bbuf;
#pragma unroll
    for (int p = 0; p < 2; ++p) {
      int it = tid + p * 128;
      int n  = it >> 2, ch = it & 3;
      async_b128(b_lds + (uint32_t)(n * LDSROW + ch * 8) * 2,
                 (uint64_t)(Bm + (size_t)(n0 + n) * ldb + k0 + ch * 8));
    }
  } else {
    // B row-major [K,N]: vector-load 8 cols, scatter-transpose into Bs[n][k]
#pragma unroll
    for (int p = 0; p < 2; ++p) {
      int it = tid + p * 128;
      int k  = it >> 3;            // 0..31
      int nb = (it & 7) * 8;       // n chunk (N always multiple of 8)
      v8bf val;
#pragma unroll
      for (int e = 0; e < 8; ++e) val[e] = (bf16_t)0.f;
      if (n0 + nb < N)
        val = *(const v8bf*)(Bm + (size_t)(k0 + k) * ldb + n0 + nb);
#pragma unroll
      for (int e = 0; e < 8; ++e) Bbuf[(size_t)(nb + e) * LDSROW + k] = val[e];
    }
  }
}

template<int EPI, int BTR>
__global__ __launch_bounds__(128)
void gemm_wmma_kernel(const bf16_t* __restrict__ A, int lda,
                      const bf16_t* __restrict__ Bm, int ldb,
                      float* __restrict__ Cf, bf16_t* __restrict__ Cb, int ldc,
                      const float* __restrict__ bias,
                      int M, int N, int K)
{
  __shared__ __align__(16) bf16_t As[2][TILE_M * LDSROW];
  __shared__ __align__(16) bf16_t Bs[2][TILE_N * LDSROW];

  const int tid  = threadIdx.x;
  const int lane = tid & 31;
  const int wave = tid >> 5;
  const int m0 = blockIdx.y * TILE_M;
  const int n0 = blockIdx.x * TILE_N;

  v8f acc[2][4];
#pragma unroll
  for (int s = 0; s < 2; ++s)
#pragma unroll
    for (int j = 0; j < 4; ++j)
#pragma unroll
      for (int e = 0; e < 8; ++e) acc[s][j][e] = 0.f;

  const int nk = K >> 5;
  stage_tile<BTR>(A, lda, Bm, ldb, m0, n0, N, 0, As[0], Bs[0], tid);

  for (int i = 0; i < nk; ++i) {
    const int cur = i & 1;
    if (i + 1 < nk) {
      stage_tile<BTR>(A, lda, Bm, ldb, m0, n0, N, (i + 1) << 5,
                      As[cur ^ 1], Bs[cur ^ 1], tid);
      if (BTR) asm volatile("s_wait_asynccnt 6" ::: "memory");
      else     asm volatile("s_wait_asynccnt 4" ::: "memory");
    } else {
      asm volatile("s_wait_asynccnt 0" ::: "memory");
    }
    __syncthreads();

    const int half = lane >> 4;
    const bf16_t* Ab = As[cur];
    const bf16_t* Bb = Bs[cur];
    v16bf af0 = load_frag(Ab + (wave * 32 +      (lane & 15)) * LDSROW, half);
    v16bf af1 = load_frag(Ab + (wave * 32 + 16 + (lane & 15)) * LDSROW, half);
#pragma unroll
    for (int j = 0; j < 4; ++j) {
      v16bf bf = load_frag(Bb + (j * 16 + (lane & 15)) * LDSROW, half);
      acc[0][j] = __builtin_amdgcn_wmma_f32_16x16x32_bf16(
          false, af0, false, bf, (short)0, acc[0][j], false, false);
      acc[1][j] = __builtin_amdgcn_wmma_f32_16x16x32_bf16(
          false, af1, false, bf, (short)0, acc[1][j], false, false);
    }
    __syncthreads();
  }

  const int half = lane >> 4;
#pragma unroll
  for (int s = 0; s < 2; ++s)
#pragma unroll
    for (int j = 0; j < 4; ++j) {
      int gn = n0 + j * 16 + (lane & 15);
      if (gn >= N) continue;
#pragma unroll
      for (int e = 0; e < 8; ++e) {
        int gm = m0 + wave * 32 + s * 16 + half * 8 + e;
        float v = acc[s][j][e];
        if (EPI == 1)      { Cf[(size_t)gm * ldc + gn] = v + bias[gn]; }
        else if (EPI == 2) { Cb[(size_t)gm * ldc + gn] = (bf16_t)tanhf(v); }
        else if (EPI == 3) { float r = v > 0.f ? v : 0.f; Cb[(size_t)gm * ldc + gn] = (bf16_t)(r * r); }
        else               { Cf[(size_t)gm * ldc + gn] = v; }
      }
    }
}

// ---------------- LayerNorm (one block per token) ----------------
__global__ __launch_bounds__(256)
void layernorm_kernel(const float* __restrict__ x, const float* __restrict__ g,
                      const float* __restrict__ b, float* __restrict__ y)
{
  const int row = blockIdx.x;
  const int tid = threadIdx.x;
  __shared__ float s_sum[256], s_sq[256];
  const float* xr = x + (size_t)row * Dn;
  float s = 0.f, q = 0.f;
  for (int i = tid; i < Dn; i += 256) { float v = xr[i]; s += v; q += v * v; }
  s_sum[tid] = s; s_sq[tid] = q;
  __syncthreads();
  for (int st = 128; st > 0; st >>= 1) {
    if (tid < st) { s_sum[tid] += s_sum[tid + st]; s_sq[tid] += s_sq[tid + st]; }
    __syncthreads();
  }
  float mean = s_sum[0] * (1.f / Dn);
  float var  = s_sq[0] * (1.f / Dn) - mean * mean;
  float rstd = rsqrtf(var + EPSn);
  for (int i = tid; i < Dn; i += 256)
    y[(size_t)row * Dn + i] = (xr[i] - mean) * rstd * g[i] + b[i];
}

// ---------------- token-shift lerp kernels (produce bf16 GEMM inputs) ----------------
__global__ __launch_bounds__(256)
void lerp_mu_kernel(const float* __restrict__ x, const float* __restrict__ mu,
                    bf16_t* __restrict__ out)
{
  size_t idx = (size_t)blockIdx.x * 256 + threadIdx.x;
  if (idx >= (size_t)Bn * Tn * Dn) return;
  int d = (int)(idx % Dn);
  size_t bt = idx / Dn;
  int t = (int)(bt % Tn);
  float xv = x[idx];
  float xp = (t > 0) ? x[idx - Dn] : 0.f;
  out[idx] = (bf16_t)(xv + (xp - xv) * mu[d]);
}

__global__ __launch_bounds__(256)
void lerp_m_kernel(const float* __restrict__ x, const float* __restrict__ m,
                   bf16_t* __restrict__ out)
{
  size_t idx = (size_t)blockIdx.x * 256 + threadIdx.x;
  if (idx >= (size_t)Bn * Tn * Dn) return;
  size_t bt = idx / Dn;
  int t = (int)(bt % Tn);
  float xv = x[idx];
  float xp = (t > 0) ? x[idx - Dn] : 0.f;
  out[idx] = (bf16_t)(xv + (xp - xv) * m[idx]);
}

// ---------------- WKV6 recurrence: one block per (b,h), 64 threads = v columns ----------------
__global__ __launch_bounds__(64)
void wkv6_kernel(const float* __restrict__ r, const float* __restrict__ k,
                 const float* __restrict__ v, const float* __restrict__ w,
                 const float* __restrict__ u, float* __restrict__ o)
{
  const int bh = blockIdx.x;
  const int b = bh / Hn, h = bh % Hn;
  const int tid = threadIdx.x;           // v column
  __shared__ float s_r[DKn], s_k[DKn], s_dec[DKn], s_u[DKn];
  float S[DKn];
#pragma unroll
  for (int i = 0; i < DKn; ++i) S[i] = 0.f;
  s_u[tid] = u[h * DKn + tid];
  const size_t base = (size_t)b * Tn * Dn + h * DKn;
  for (int t = 0; t < Tn; ++t) {
    const size_t off = base + (size_t)t * Dn;
    __syncthreads();
    s_r[tid]   = r[off + tid];
    s_k[tid]   = k[off + tid];
    s_dec[tid] = __expf(-__expf(w[off + tid]));   // decay = exp(-exp(w))
    __syncthreads();
    float vt = v[off + tid];
    float acc = 0.f;
#pragma unroll
    for (int i = 0; i < DKn; ++i) {
      float kv = s_k[i] * vt;
      acc += s_r[i] * (S[i] + s_u[i] * kv);
      S[i] = s_dec[i] * S[i] + kv;
    }
    o[off + tid] = acc;
  }
}

// ---------------- per-head GroupNorm * swish(g) -> bf16 ----------------
__global__ __launch_bounds__(256)
void gn_gate_kernel(const float* __restrict__ o, const float* __restrict__ g,
                    const float* __restrict__ gg, const float* __restrict__ gb,
                    bf16_t* __restrict__ out)
{
  const int row = blockIdx.x;
  const int tid = threadIdx.x;           // 256 threads, 4 elems each; head = tid/16
  __shared__ float ps[256], pq[256];
  __shared__ float hm[Hn], hr[Hn];
  const float* orow = o + (size_t)row * Dn;
  float vals[4];
  float s = 0.f, q = 0.f;
#pragma unroll
  for (int j = 0; j < 4; ++j) { vals[j] = orow[tid * 4 + j]; s += vals[j]; q += vals[j] * vals[j]; }
  ps[tid] = s; pq[tid] = q;
  __syncthreads();
  if (tid < Hn) {
    float a = 0.f, c = 0.f;
    for (int i = 0; i < 16; ++i) { a += ps[tid * 16 + i]; c += pq[tid * 16 + i]; }
    float m = a * (1.f / DVn);
    float v = c * (1.f / DVn) - m * m;
    hm[tid] = m; hr[tid] = rsqrtf(v + EPSn);
  }
  __syncthreads();
  const int h = tid / 16;
#pragma unroll
  for (int j = 0; j < 4; ++j) {
    int d = tid * 4 + j;
    float n  = (vals[j] - hm[h]) * hr[h] * gg[d] + gb[d];
    float gv = g[(size_t)row * Dn + d];
    float sw = gv / (1.f + __expf(-gv));           // g * sigmoid(g)
    out[(size_t)row * Dn + d] = (bf16_t)(n * sw);
  }
}

// ---------------- elementwise helpers ----------------
__global__ __launch_bounds__(256)
void f2bf_kernel(const float* __restrict__ x, bf16_t* __restrict__ y, size_t n)
{
  size_t i = (size_t)blockIdx.x * 256 + threadIdx.x;
  if (i < n) y[i] = (bf16_t)x[i];
}

__global__ __launch_bounds__(256)
void add_inplace_kernel(float* __restrict__ a, const float* __restrict__ b, size_t n)
{
  size_t i = (size_t)blockIdx.x * 256 + threadIdx.x;
  if (i < n) a[i] += b[i];
}

__global__ __launch_bounds__(256)
void final_kernel(const float* __restrict__ res, const float* __restrict__ rec,
                  const float* __restrict__ kv, float* __restrict__ out, size_t n)
{
  size_t i = (size_t)blockIdx.x * 256 + threadIdx.x;
  if (i < n) out[i] = res[i] + kv[i] / (1.f + __expf(-rec[i]));
}

// ---------------- host orchestration ----------------
extern "C" void kernel_launch(void* const* d_in, const int* in_sizes, int n_in,
                              void* d_out, int out_size, void* d_ws, size_t ws_size,
                              hipStream_t stream)
{
  (void)in_sizes; (void)n_in; (void)out_size; (void)ws_size;
  const float* IN[45];
  for (int i = 0; i < 45; ++i) IN[i] = (const float*)d_in[i];

  const float *query = IN[0], *keyval = IN[1];
  const float *ln_pre_g = IN[2], *ln_pre_b = IN[3];
  const float *ln_q_g   = IN[4], *ln_q_b   = IN[5];
  const float *ln_kv_g  = IN[6], *ln_kv_b  = IN[7];
  const float *ln_ffn_g = IN[8], *ln_ffn_b = IN[9];
  const float *ffn_mu_k = IN[40], *ffn_key_w = IN[41], *ffn_val_w = IN[42];
  const float *ffn_mu_r = IN[43], *ffn_rec_w = IN[44];

  const size_t S = (size_t)BTn * Dn;

  // workspace allocator
  char* p = (char*)d_ws;
  auto alignp = [&]() { p = (char*)(((uintptr_t)p + 255) & ~(uintptr_t)255); };
  auto allocF = [&](size_t n) { alignp(); float*  r = (float*)p;  p += n * sizeof(float);  return r; };
  auto allocH = [&](size_t n) { alignp(); bf16_t* r = (bf16_t*)p; p += n * sizeof(bf16_t); return r; };

  auto conv = [&](const float* src, size_t n) {
    bf16_t* dst = allocH(n);
    f2bf_kernel<<<dim3((unsigned)((n + 255) / 256)), 256, 0, stream>>>(src, dst, n);
    return dst;
  };

  struct AttnW {
    const float *x_mu, *x_bias, *w_bias, *u, *gn_g, *gn_b;
    bf16_t *x_w1, *x_w2, *r_w, *k_w, *v_w, *g_w, *w_a, *w_b, *o_w;
  };
  auto load_attn = [&](int base) {
    AttnW w;
    w.x_mu   = IN[base + 0];
    w.x_w1   = conv(IN[base + 1], (size_t)Dn * 5 * Rn);
    w.x_w2   = conv(IN[base + 2], (size_t)Dn * 5 * Rn);
    w.x_bias = IN[base + 3];
    w.r_w    = conv(IN[base + 4], (size_t)Dn * Dn);
    w.k_w    = conv(IN[base + 5], (size_t)Dn * Dn);
    w.v_w    = conv(IN[base + 6], (size_t)Dn * Dn);
    w.g_w    = conv(IN[base + 7], (size_t)Dn * Dn);
    w.w_a    = conv(IN[base + 8], (size_t)Dn * GLRn);
    w.w_b    = conv(IN[base + 9], (size_t)GLRn * Dn);
    w.w_bias = IN[base + 10];
    w.u      = IN[base + 11];
    w.gn_g   = IN[base + 12];
    w.gn_b   = IN[base + 13];
    w.o_w    = conv(IN[base + 14], (size_t)Dn * Dn);
    return w;
  };
  AttnW sa = load_attn(10);
  AttnW ca = load_attn(25);
  bf16_t* key_h = conv(ffn_key_w, (size_t)Dn * INTERn);
  bf16_t* val_h = conv(ffn_val_w, (size_t)INTERn * Dn);
  bf16_t* rec_h = conv(ffn_rec_w, (size_t)Dn * Dn);

  // activation scratch
  float *res = allocF(S), *x0 = allocF(S), *x1 = allocF(S), *kvb = allocF(S);
  float *msl = allocF(S), *rb = allocF(S), *kb = allocF(S), *vb = allocF(S);
  float *wb_ = allocF(S), *gb = allocF(S), *ob = allocF(S);
  bf16_t *lerp = allocH(S), *gated = allocH(S);
  bf16_t *low  = allocH((size_t)BTn * 5 * Rn);
  bf16_t *wlow = allocH((size_t)BTn * GLRn);
  bf16_t *kk   = allocH((size_t)BTn * INTERn);

  auto gemm = [&](int epi, int btr, const bf16_t* A, int lda, const bf16_t* Bm, int ldb,
                  float* Cf, bf16_t* Cb, int ldc, const float* bias, int M, int N, int K) {
    dim3 grid((N + TILE_N - 1) / TILE_N, (M + TILE_M - 1) / TILE_M), blk(128);
    if (btr) {
      gemm_wmma_kernel<1, 1><<<grid, blk, 0, stream>>>(A, lda, Bm, ldb, Cf, Cb, ldc, bias, M, N, K);
    } else if (epi == 0) {
      gemm_wmma_kernel<0, 0><<<grid, blk, 0, stream>>>(A, lda, Bm, ldb, Cf, Cb, ldc, bias, M, N, K);
    } else if (epi == 1) {
      gemm_wmma_kernel<1, 0><<<grid, blk, 0, stream>>>(A, lda, Bm, ldb, Cf, Cb, ldc, bias, M, N, K);
    } else if (epi == 2) {
      gemm_wmma_kernel<2, 0><<<grid, blk, 0, stream>>>(A, lda, Bm, ldb, Cf, Cb, ldc, bias, M, N, K);
    } else {
      gemm_wmma_kernel<3, 0><<<grid, blk, 0, stream>>>(A, lda, Bm, ldb, Cf, Cb, ldc, bias, M, N, K);
    }
  };

  const dim3 gE((unsigned)((S + 255) / 256)), bE(256);

  auto attn = [&](const float* xq, const float* xkv, bool self_mode, const AttnW& w, float* outbuf) {
    lerp_mu_kernel<<<gE, bE, 0, stream>>>(xq, w.x_mu, lerp);
    gemm(2, 0, lerp, Dn, w.x_w1, 5 * Rn, nullptr, low, 5 * Rn, nullptr, BTn, 5 * Rn, Dn);
    auto slice = [&](const float* xs, int n, const bf16_t* W, int Nw, float* Cf) {
      gemm(1, 1, low + n * Rn, 5 * Rn, w.x_w2 + n * Rn, 5 * Rn,
           msl, nullptr, Dn, w.x_bias + n * Dn, BTn, Dn, Rn);
      lerp_m_kernel<<<gE, bE, 0, stream>>>(xs, msl, lerp);
      gemm(0, 0, lerp, Dn, W, Nw, Cf, nullptr, Nw, nullptr, BTn, Nw, Dn);
    };
    slice(xq, 0, w.r_w, Dn, rb);   // r
    slice(xq, 4, w.g_w, Dn, gb);   // g
    if (!self_mode) {              // mus basis from xkv for w/k/v
      lerp_mu_kernel<<<gE, bE, 0, stream>>>(xkv, w.x_mu, lerp);
      gemm(2, 0, lerp, Dn, w.x_w1, 5 * Rn, nullptr, low, 5 * Rn, nullptr, BTn, 5 * Rn, Dn);
    }
    // w path (n=1): tanh(@w_a) @ w_b + w_bias
    gemm(1, 1, low + 1 * Rn, 5 * Rn, w.x_w2 + 1 * Rn, 5 * Rn,
         msl, nullptr, Dn, w.x_bias + 1 * Dn, BTn, Dn, Rn);
    lerp_m_kernel<<<gE, bE, 0, stream>>>(xkv, msl, lerp);
    gemm(2, 0, lerp, Dn, w.w_a, GLRn, nullptr, wlow, GLRn, nullptr, BTn, GLRn, Dn);
    gemm(1, 0, wlow, GLRn, w.w_b, Dn, wb_, nullptr, Dn, w.w_bias, BTn, Dn, GLRn);
    slice(xkv, 2, w.k_w, Dn, kb);  // k
    slice(xkv, 3, w.v_w, Dn, vb);  // v
    wkv6_kernel<<<Bn * Hn, DVn, 0, stream>>>(rb, kb, vb, wb_, w.u, ob);
    gn_gate_kernel<<<BTn, 256, 0, stream>>>(ob, gb, w.gn_g, w.gn_b, gated);
    gemm(0, 0, gated, Dn, w.o_w, Dn, outbuf, nullptr, Dn, nullptr, BTn, Dn, Dn);
  };

  // ---------- forward ----------
  layernorm_kernel<<<BTn, 256, 0, stream>>>(query,  ln_pre_g, ln_pre_b, res);
  layernorm_kernel<<<BTn, 256, 0, stream>>>(res,    ln_q_g,   ln_q_b,   x0);
  layernorm_kernel<<<BTn, 256, 0, stream>>>(keyval, ln_kv_g,  ln_kv_b,  kvb);

  attn(x0, x0,  true,  sa, x1);   // self time-mix
  attn(x1, kvb, false, ca, x0);   // cross time-mix

  add_inplace_kernel<<<gE, bE, 0, stream>>>(res, x0, S);            // res = q + res
  layernorm_kernel<<<BTn, 256, 0, stream>>>(res, ln_ffn_g, ln_ffn_b, x1);

  // FFN
  lerp_mu_kernel<<<gE, bE, 0, stream>>>(x1, ffn_mu_k, lerp);
  gemm(3, 0, lerp, Dn, key_h, INTERn, nullptr, kk, INTERn, nullptr, BTn, INTERn, Dn); // sqrelu
  lerp_mu_kernel<<<gE, bE, 0, stream>>>(x1, ffn_mu_r, lerp);
  gemm(0, 0, lerp, Dn, rec_h, Dn, rb, nullptr, Dn, nullptr, BTn, Dn, Dn);             // rec
  gemm(0, 0, kk, INTERn, val_h, Dn, kb, nullptr, Dn, nullptr, BTn, Dn, INTERn);       // kk @ val_w

  final_kernel<<<gE, bE, 0, stream>>>(res, rb, kb, (float*)d_out, S);
}
